// Attention_20624432955701
// MI455X (gfx1250) — compile-verified, matrix operand order checked by reading, not run on hardware
//
#include <hip/hip_runtime.h>
#include <hip/hip_bf16.h>

typedef __attribute__((ext_vector_type(16))) _Float16 v16h;
typedef __attribute__((ext_vector_type(8)))  _Float16 v8h;
typedef __attribute__((ext_vector_type(8)))  float    v8f;

namespace {
constexpr int SEQ  = 2048;   // sequence length
constexpr int DH   = 64;     // head dim
constexpr int NX   = 1024;   // H * DH
constexpr int HN   = 16;     // heads
constexpr int BB   = 2;      // batch
constexpr int MQ   = 32;     // query rows per workgroup
constexpr int KCH  = 256;    // keys staged per chunk (16 waves * 16)
constexpr int LDP  = 72;     // f16 row stride (64 + 8 pad, keeps 16B alignment)
constexpr int NTHREADS = 512;
constexpr float NEG_BIG = -1e10f;

// LDS layout (bytes):
//  qlds   : MQ  * LDP * 2          =   4608
//  kprlds : KCH * LDP * 2          =  36864
//  sc     : MQ * SEQ * 4           = 262144
//  acc    : MQ * DH * 4            =   8192
//  invs   : MQ * 4                 =    128
constexpr size_t F16_BYTES = (size_t)(MQ + KCH) * LDP * 2;   // 41472 (multiple of 256)
constexpr size_t SMEM_BYTES = F16_BYTES + (size_t)MQ * SEQ * 4 + (size_t)MQ * DH * 4 + MQ * 4;
} // namespace

// Load a 16x32 f16 fragment (A or B role) from an LDS tile with row stride LDP.
// Per ISA 16-bit layout: lane(0..15) holds row/col = lane&15,
// halves j=0..7  -> K = kbase + 8*hi + j
// halves j=8..15 -> K = kbase + 16 + 8*hi + (j-8)     (hi = lane>>4)
__device__ inline v16h load_frag_lds(const _Float16* __restrict__ base, int row,
                                     int kbase, int hi) {
  union { v16h v; v8h h2[2]; } u;
  const _Float16* p = base + row * LDP;
  u.h2[0] = *(const v8h*)(p + kbase + 8 * hi);
  u.h2[1] = *(const v8h*)(p + kbase + 16 + 8 * hi);
  return u.v;
}

__global__ __launch_bounds__(NTHREADS)
void attn_wmma_kernel(const float* __restrict__ qG, const float* __restrict__ kG,
                      const float* __restrict__ vG, const float* __restrict__ krG,
                      float* __restrict__ outA, float* __restrict__ outAtt) {
  extern __shared__ char smem[];
  _Float16* qlds   = (_Float16*)smem;                 // [MQ][LDP]
  _Float16* kprlds = qlds + MQ * LDP;                 // [KCH][LDP]
  float*    sc     = (float*)(smem + F16_BYTES);      // [MQ][SEQ]  exp-scores
  float*    acc    = sc + (size_t)MQ * SEQ;           // [MQ][DH]
  float*    invs   = acc + MQ * DH;                   // [MQ]

  const int qt = blockIdx.x;          // 0..63
  const int h  = blockIdx.y;          // 0..15
  const int b  = blockIdx.z;          // 0..1
  const int tid  = threadIdx.x;
  const int lane = tid & 31;
  const int wave = tid >> 5;          // 0..15
  const int hi   = lane >> 4;
  const int ln   = lane & 15;
  const int q0   = qt * MQ;

  const size_t rowOff = (size_t)b * SEQ * NX + (size_t)h * DH;
  const float* qg  = qG  + rowOff;    // element (s, c): qg + s*NX + c
  const float* kg  = kG  + rowOff;
  const float* krg = krG + rowOff;
  const float* vg  = vG  + rowOff;

  // ---- phase 1: stage Q tile (f32 -> f16) into LDS; zero accumulators ----
  for (int i = tid; i < MQ * DH; i += NTHREADS) {
    int r = i >> 6, c = i & 63;
    qlds[r * LDP + c] = (_Float16)qg[(size_t)(q0 + r) * NX + c];
    acc[i] = 0.0f;
  }
  __syncthreads();

  // Preload the 4 constant A fragments (2 m-tiles x 2 K-halves of d=64)
  v16h afr[2][2];
  #pragma unroll
  for (int mt = 0; mt < 2; ++mt)
    #pragma unroll
    for (int kk = 0; kk < 2; ++kk)
      afr[mt][kk] = load_frag_lds(qlds, mt * 16 + ln, kk * 32, hi);

  // ---- phase 2: scores = Q (K+Kr)^T / sqrt(d), causal mask, into LDS ----
  for (int ch = 0; ch < SEQ; ch += KCH) {
    __syncthreads();   // previous chunk's WMMA reads of kprlds complete
    // cooperative load of (K + Kr) chunk, f32 add, f16 convert
    for (int i = tid; i < KCH * 16; i += NTHREADS) {
      int r = i >> 4, c4 = i & 15;
      float4 kv = *(const float4*)(kg  + (size_t)(ch + r) * NX + c4 * 4);
      float4 rv = *(const float4*)(krg + (size_t)(ch + r) * NX + c4 * 4);
      _Float16* dst = kprlds + r * LDP + c4 * 4;
      dst[0] = (_Float16)(kv.x + rv.x);
      dst[1] = (_Float16)(kv.y + rv.y);
      dst[2] = (_Float16)(kv.z + rv.z);
      dst[3] = (_Float16)(kv.w + rv.w);
    }
    __syncthreads();

    const int kb = wave * 16;                 // this wave's 16 keys in chunk
    v16h bfr[2];
    #pragma unroll
    for (int kk = 0; kk < 2; ++kk)
      bfr[kk] = load_frag_lds(kprlds, kb + ln, kk * 32, hi);

    #pragma unroll
    for (int mt = 0; mt < 2; ++mt) {
      v8f c = {};
      c = __builtin_amdgcn_wmma_f32_16x16x32_f16(false, afr[mt][0], false, bfr[0],
                                                 (short)0, c, false, false);
      c = __builtin_amdgcn_wmma_f32_16x16x32_f16(false, afr[mt][1], false, bfr[1],
                                                 (short)0, c, false, false);
      const int col = ch + kb + ln;           // global key index
      #pragma unroll
      for (int r = 0; r < 8; ++r) {
        int mrow = mt * 16 + r + 8 * hi;      // local q row 0..31
        float val = c[r] * 0.125f;            // 1/sqrt(64)
        if (col > q0 + mrow) val = NEG_BIG;   // causal mask
        sc[(size_t)mrow * SEQ + col] = val;
      }
    }
  }
  __syncthreads();

  // ---- phase 3: exact softmax per row (wave handles 2 rows) ----
  #pragma unroll
  for (int rr = 0; rr < 2; ++rr) {
    const int row = wave * 2 + rr;
    float* srow = sc + (size_t)row * SEQ;
    float m = -3.0e38f;
    for (int c = lane; c < SEQ; c += 32) m = fmaxf(m, srow[c]);
    #pragma unroll
    for (int off = 16; off; off >>= 1) m = fmaxf(m, __shfl_xor(m, off, 32));
    float s = 0.0f;
    for (int c = lane; c < SEQ; c += 32) {
      float e = __expf(srow[c] - m);
      srow[c] = e;                            // keep unnormalized exp in LDS
      s += e;
    }
    #pragma unroll
    for (int off = 16; off; off >>= 1) s += __shfl_xor(s, off, 32);
    if (lane == 0) invs[row] = 1.0f / s;
  }
  __syncthreads();

  // ---- phase 4: stream normalized att to global (float4 stores) ----
  {
    float* attBase = outAtt + (((size_t)b * HN + h) * SEQ + q0) * SEQ;
    for (int i = tid; i < MQ * (SEQ / 4); i += NTHREADS) {
      int r = i / (SEQ / 4), c4 = i % (SEQ / 4);
      float is = invs[r];
      float4 e = *(const float4*)(sc + (size_t)r * SEQ + c4 * 4);
      float4 o = make_float4(e.x * is, e.y * is, e.z * is, e.w * is);
      *(float4*)(attBase + (size_t)r * SEQ + c4 * 4) = o;
    }
  }

  // ---- phase 5: a = exp(scores) @ V ; each wave owns a 128-key K slice ----
  {
    v8f co[2][4];
    #pragma unroll
    for (int mt = 0; mt < 2; ++mt)
      #pragma unroll
      for (int nt = 0; nt < 4; ++nt) co[mt][nt] = (v8f){0,0,0,0,0,0,0,0};

    const int ks0 = wave * 128;
    for (int kk = ks0; kk < ks0 + 128; kk += 32) {
      const int kA = kk + 8 * hi;
      const int kB = kk + 16 + 8 * hi;
      v16h ae[2];
      #pragma unroll
      for (int mt = 0; mt < 2; ++mt) {
        const float* srow = sc + (size_t)(mt * 16 + ln) * SEQ;
        union { v16h v; } u;
        #pragma unroll
        for (int j = 0; j < 8; ++j) {
          u.v[j]     = (_Float16)srow[kA + j];
          u.v[8 + j] = (_Float16)srow[kB + j];
        }
        ae[mt] = u.v;
      }
      #pragma unroll
      for (int nt = 0; nt < 4; ++nt) {
        const float* vcol = vg + nt * 16 + ln;   // lanes 0..15 coalesce 64B
        union { v16h v; } u;
        #pragma unroll
        for (int j = 0; j < 8; ++j) {
          u.v[j]     = (_Float16)vcol[(size_t)(kA + j) * NX];
          u.v[8 + j] = (_Float16)vcol[(size_t)(kB + j) * NX];
        }
        #pragma unroll
        for (int mt = 0; mt < 2; ++mt)
          co[mt][nt] = __builtin_amdgcn_wmma_f32_16x16x32_f16(
              false, ae[mt], false, u.v, (short)0, co[mt][nt], false, false);
      }
    }
    // reduce 16 wave-partials with LDS float atomics (ds_add_f32)
    #pragma unroll
    for (int mt = 0; mt < 2; ++mt)
      #pragma unroll
      for (int nt = 0; nt < 4; ++nt)
        #pragma unroll
        for (int r = 0; r < 8; ++r) {
          int mrow = mt * 16 + r + 8 * hi;
          int c = nt * 16 + ln;
          __hip_atomic_fetch_add(&acc[mrow * DH + c], co[mt][nt][r],
                                 __ATOMIC_RELAXED, __HIP_MEMORY_SCOPE_WORKGROUP);
        }
  }
  __syncthreads();

  // ---- phase 6: write a, merged heads: out[b][s][h*64 + c] ----
  {
    float* aBase = outA + ((size_t)b * SEQ + q0) * NX + (size_t)h * DH;
    for (int i = tid; i < MQ * DH; i += NTHREADS) {
      int r = i >> 6, c = i & 63;
      aBase[(size_t)r * NX + c] = acc[i] * invs[r];
    }
  }
}

extern "C" void kernel_launch(void* const* d_in, const int* in_sizes, int n_in,
                              void* d_out, int out_size, void* d_ws, size_t ws_size,
                              hipStream_t stream) {
  (void)in_sizes; (void)n_in; (void)out_size; (void)d_ws; (void)ws_size;
  const float* q  = (const float*)d_in[0];
  const float* k  = (const float*)d_in[1];
  const float* v  = (const float*)d_in[2];
  const float* kr = (const float*)d_in[3];
  float* outA   = (float*)d_out;                                   // [B,S,NX]
  float* outAtt = outA + (size_t)BB * SEQ * NX;                    // [B,H,S,S]

  dim3 grid(SEQ / MQ, HN, BB);   // (64, 16, 2)
  dim3 block(NTHREADS);
  attn_wmma_kernel<<<grid, block, SMEM_BYTES, stream>>>(q, k, v, kr, outA, outAtt);
}
